// DeepSeekPINN_57054345560342
// MI455X (gfx1250) — matrix operand order repaired and checked by reading, not compiled, not run on hardware
//
#include <hip/hip_runtime.h>

#define LPAD   72          // padded LDS row stride (bf16 elements) -> 144B, bank-spread
#define LAYERS 3

typedef __bf16 bf16;
typedef __attribute__((ext_vector_type(16))) __bf16 v16bf;
typedef __attribute__((ext_vector_type(8)))  __bf16 v8bf;
typedef __attribute__((ext_vector_type(8)))  float   v8f;

// tanh(x) = 1 - 2/(exp(2x)+1); use raw v_rcp_f32 instead of IEEE division
// (avoids the div_scale/div_fmas/div_fixup expansion seen in round-1 asm).
static __device__ __forceinline__ float fast_tanh(float x) {
  float e = __expf(2.0f * x);
  return fmaf(-2.0f, __builtin_amdgcn_rcpf(e + 1.0f), 1.0f);
}

static __device__ __forceinline__ v16bf cat16(v8bf lo, v8bf hi) {
  return __builtin_shufflevector(lo, hi, 0,1,2,3,4,5,6,7,8,9,10,11,12,13,14,15);
}

// A-fragment (16x32 bf16, M=row, K-major per ISA layout):
// lanes 0-15: V0-3 = K 0..7, V4-7 = K 16..23 ; lanes 16-31: +8 on both K groups
static __device__ __forceinline__ v16bf load_a_frag(const bf16* hs, int lane, int k0) {
  const int M = lane & 15;
  const int s = (lane >> 4) << 3;                 // 0 or 8
  const bf16* row = hs + M * LPAD;
  v8bf lo = *(const v8bf*)(row + k0 + s);
  v8bf hi = *(const v8bf*)(row + k0 + 16 + s);
  return cat16(lo, hi);
}

// B-fragment (32x16 bf16): lane = column N, lanes 0-15 hold K 0..7,16..23 (kb=k0),
// lanes 16-31 hold K 8..15,24..31 (kb=k0+8).  Weights stored transposed [out][in] so K is contiguous.
static __device__ __forceinline__ v16bf load_b_frag(const bf16* wt, int lane, int n0, int k0) {
  const int N  = (n0 << 4) + (lane & 15);
  const int kb = k0 + ((lane >> 4) << 3);
  const bf16* p = wt + N * LPAD + kb;
  v8bf lo = *(const v8bf*)(p);
  v8bf hi = *(const v8bf*)(p + 16);
  return cat16(lo, hi);
}

static __device__ __forceinline__ v8f wmma_bf16(v16bf a, v16bf b, v8f c) {
  return __builtin_amdgcn_wmma_f32_16x16x32_bf16(false, a, false, b, (short)0, c, false, false);
}

// ---- prep: fp32 weights -> bf16, transposed to [layer][expert][mat][out][in] ----
__global__ __launch_bounds__(256)
void DeepSeekPINN_prep_kernel(const float* __restrict__ sW1, const float* __restrict__ sW2,
                              const float* __restrict__ rW1, const float* __restrict__ rW2,
                              bf16* __restrict__ wgt)
{
  int idx = blockIdx.x * 256 + threadIdx.x;
  if (idx >= LAYERS * 6 * 2 * 64 * 64) return;
  const int in   = idx & 63;
  const int outd = (idx >> 6) & 63;
  const int m    = (idx >> 12) & 1;
  const int le   = idx >> 13;          // l*6 + e
  const int e    = le % 6;
  const int l    = le / 6;
  float v;
  if (e < 2) {
    const float* W = m ? sW2 : sW1;
    v = W[((l * 2 + e) * 64 + in) * 64 + outd];
  } else {
    const float* W = m ? rW2 : rW1;
    v = W[((l * 4 + (e - 2)) * 64 + in) * 64 + outd];
  }
  wgt[idx] = (bf16)v;
}

// ---- fused 3-layer MoE PINN forward ----
__global__ __launch_bounds__(256)
void DeepSeekPINN_moe_kernel(
    const float* __restrict__ x, const float* __restrict__ t,
    const float* __restrict__ W_in, const float* __restrict__ b_in,
    const float* __restrict__ sb1, const float* __restrict__ sb2,
    const float* __restrict__ rb1, const float* __restrict__ rb2,
    const float* __restrict__ routW, const float* __restrict__ routb,
    const float* __restrict__ W_out, const float* __restrict__ b_out,
    const bf16* __restrict__ wgt,
    float* __restrict__ out)
{
  extern __shared__ __align__(16) char dynsmem[];
  bf16* sW = (bf16*)dynsmem;                 // 12 mats * 64 * LPAD
  bf16* sH = sW + 12 * 64 * LPAD;            // 8 waves * 16 * LPAD

  const int tid  = threadIdx.x;
  const int wave = tid >> 5;
  const int lane = tid & 31;
  const int col  = lane & 15;                // N index within 16-col tile (C layout)
  const int mhi  = (lane >> 4) << 3;         // row offset: 0 (lanes 0-15) / 8 (lanes 16-31)
  bf16* hs = sH + wave * 16 * LPAD;
  const int tokBase = blockIdx.x * 128 + (wave << 4);

  // ---- input embedding: h = tanh([x t] @ W_in + b_in), kept in WMMA C-layout ----
  v8f h[4];
  float xv[8], tv[8];
#pragma unroll
  for (int r = 0; r < 8; ++r) {
    xv[r] = x[tokBase + mhi + r];
    tv[r] = t[tokBase + mhi + r];
  }
#pragma unroll
  for (int n = 0; n < 4; ++n) {
    const int d = (n << 4) + col;
    const float w0 = W_in[d], w1 = W_in[64 + d], bb = b_in[d];
#pragma unroll
    for (int r = 0; r < 8; ++r)
      h[n][r] = fast_tanh(fmaf(xv[r], w0, fmaf(tv[r], w1, bb)));
  }

  for (int l = 0; l < LAYERS; ++l) {
    // ---- stage this layer's bf16 transposed weights into LDS ----
    __syncthreads();
    {
      const bf16* src = wgt + l * (12 * 64 * 64);
      for (int c = tid; c < 12 * 64 * 8; c += 256) {       // 16B chunks
        const int mat = c >> 9, rem = c & 511, row = rem >> 3, ch = rem & 7;
        v8bf v = *(const v8bf*)(src + ((mat << 6) + row) * 64 + (ch << 3));
        *(v8bf*)(sW + mat * (64 * LPAD) + row * LPAD + (ch << 3)) = v;
      }
    }
    __syncthreads();

    // ---- stage h (bf16) and build A fragments ----
#pragma unroll
    for (int r = 0; r < 8; ++r) {
      bf16* row = hs + (mhi + r) * LPAD;
      row[col]      = (bf16)h[0][r];
      row[col + 16] = (bf16)h[1][r];
      row[col + 32] = (bf16)h[2][r];
      row[col + 48] = (bf16)h[3][r];
    }
    v16bf ah0 = load_a_frag(hs, lane, 0);
    v16bf ah1 = load_a_frag(hs, lane, 32);

    // ---- router: logits -> softmax -> top-2 scatter mask (not renormalized) ----
    float rwgt[8][4];
    {
      float rw[4][4], rb[4];
#pragma unroll
      for (int n = 0; n < 4; ++n)
#pragma unroll
        for (int e = 0; e < 4; ++e)
          rw[n][e] = routW[(l * 64 + (n << 4) + col) * 4 + e];
#pragma unroll
      for (int e = 0; e < 4; ++e) rb[e] = routb[l * 4 + e];
#pragma unroll
      for (int r = 0; r < 8; ++r) {
        float ex[4]; float mx = -1e30f;
#pragma unroll
        for (int e = 0; e < 4; ++e) {
          float p = h[0][r]*rw[0][e] + h[1][r]*rw[1][e] + h[2][r]*rw[2][e] + h[3][r]*rw[3][e];
          p += __shfl_xor(p, 1);
          p += __shfl_xor(p, 2);
          p += __shfl_xor(p, 4);
          p += __shfl_xor(p, 8);                 // sum over the 16 N-lanes
          ex[e] = p + rb[e];
          mx = fmaxf(mx, ex[e]);
        }
        float s = 0.f;
#pragma unroll
        for (int e = 0; e < 4; ++e) { ex[e] = __expf(ex[e] - mx); s += ex[e]; }
        const float inv = __builtin_amdgcn_rcpf(s);
#pragma unroll
        for (int e = 0; e < 4; ++e) {
          int cnt = 0;
#pragma unroll
          for (int e2 = 0; e2 < 4; ++e2)
            cnt += (e2 != e) && (ex[e2] > ex[e] || (ex[e2] == ex[e] && e2 < e));
          rwgt[r][e] = (cnt < 2) ? ex[e] * inv : 0.0f;   // top-2 of 4, lower index wins ties
        }
      }
    }

    v8f acc[4] = {};

    // ---- experts: 0..1 shared (weight 1), 2..5 routed (router weight) ----
#pragma unroll
    for (int e = 0; e < 6; ++e) {
      const bf16* w1p = sW + (e * 2 + 0) * (64 * LPAD);
      const bf16* w2p = sW + (e * 2 + 1) * (64 * LPAD);
      const float* b1 = (e < 2) ? sb1 + (l * 2 + e) * 64 : rb1 + (l * 4 + (e - 2)) * 64;
      const float* b2 = (e < 2) ? sb2 + (l * 2 + e) * 64 : rb2 + (l * 4 + (e - 2)) * 64;

      // mid = tanh(h @ W1 + b1), staged back to LDS for the second GEMM's A side
#pragma unroll
      for (int n = 0; n < 4; ++n) {
        v8f c = {};
        c = wmma_bf16(ah0, load_b_frag(w1p, lane, n, 0),  c);
        c = wmma_bf16(ah1, load_b_frag(w1p, lane, n, 32), c);
        const float b1v = b1[(n << 4) + col];
#pragma unroll
        for (int r = 0; r < 8; ++r)
          hs[(mhi + r) * LPAD + (n << 4) + col] = (bf16)fast_tanh(c[r] + b1v);
      }
      v16bf as0 = load_a_frag(hs, lane, 0);
      v16bf as1 = load_a_frag(hs, lane, 32);

#pragma unroll
      for (int n = 0; n < 4; ++n) {
        v8f c = {};
        c = wmma_bf16(as0, load_b_frag(w2p, lane, n, 0),  c);
        c = wmma_bf16(as1, load_b_frag(w2p, lane, n, 32), c);
        const float b2v = b2[(n << 4) + col];
        if (e < 2) {
#pragma unroll
          for (int r = 0; r < 8; ++r) acc[n][r] += c[r] + b2v;
        } else {
#pragma unroll
          for (int r = 0; r < 8; ++r) acc[n][r] += rwgt[r][e - 2] * (c[r] + b2v);
        }
      }
    }

    // ---- residual + tanh ----
#pragma unroll
    for (int n = 0; n < 4; ++n)
#pragma unroll
      for (int r = 0; r < 8; ++r)
        h[n][r] = fast_tanh(h[n][r] + acc[n][r]);
  }

  // ---- output head: out = h @ W_out + b_out (cross-lane reduce over 64 dims) ----
  const float wo0 = W_out[col], wo1 = W_out[16 + col], wo2 = W_out[32 + col], wo3 = W_out[48 + col];
  const float bo = b_out[0];
#pragma unroll
  for (int r = 0; r < 8; ++r) {
    float p = h[0][r]*wo0 + h[1][r]*wo1 + h[2][r]*wo2 + h[3][r]*wo3;
    p += __shfl_xor(p, 1);
    p += __shfl_xor(p, 2);
    p += __shfl_xor(p, 4);
    p += __shfl_xor(p, 8);
    if (col == 0) out[tokBase + mhi + r] = p + bo;
  }
}

extern "C" void kernel_launch(void* const* d_in, const int* in_sizes, int n_in,
                              void* d_out, int out_size, void* d_ws, size_t ws_size,
                              hipStream_t stream) {
  const float* x     = (const float*)d_in[0];
  const float* t     = (const float*)d_in[1];
  const float* W_in  = (const float*)d_in[2];
  const float* b_in  = (const float*)d_in[3];
  const float* sW1   = (const float*)d_in[4];
  const float* sb1   = (const float*)d_in[5];
  const float* sW2   = (const float*)d_in[6];
  const float* sb2   = (const float*)d_in[7];
  const float* rW1   = (const float*)d_in[8];
  const float* rb1   = (const float*)d_in[9];
  const float* rW2   = (const float*)d_in[10];
  const float* rb2   = (const float*)d_in[11];
  const float* routW = (const float*)d_in[12];
  const float* routb = (const float*)d_in[13];
  const float* W_out = (const float*)d_in[14];
  const float* b_out = (const float*)d_in[15];
  float* outp = (float*)d_out;
  bf16* wgt = (bf16*)d_ws;                       // 294,912 B of bf16 weights
  const int N = in_sizes[0];

  const int wtot = LAYERS * 6 * 2 * 64 * 64;
  DeepSeekPINN_prep_kernel<<<(wtot + 255) / 256, 256, 0, stream>>>(sW1, sW2, rW1, rW2, wgt);

  const size_t lds = (size_t)(12 * 64 * LPAD + 8 * 16 * LPAD) * sizeof(bf16);  // ~126 KB
  DeepSeekPINN_moe_kernel<<<N / 128, 256, lds, stream>>>(
      x, t, W_in, b_in, sb1, sb2, rb1, rb2, routW, routb, W_out, b_out, wgt, outp);
}
